// GroupedCrossAttention_76072460747235
// MI455X (gfx1250) — matrix operand order repaired and checked by reading, not compiled, not run on hardware
//
#include <hip/hip_runtime.h>
#include <hip/hip_bf16.h>

typedef __attribute__((ext_vector_type(16))) _Float16 v16h;
typedef __attribute__((ext_vector_type(8)))  float    v8f;

constexpr int Bn  = 4;
constexpr int SQn = 1024;
constexpr int SKn = 2048;
constexpr int Dn  = 1024;
constexpr int Hn  = 16;
constexpr int Gn  = 4;
constexpr int HDn = 64;   // Dn / Hn
constexpr int GDn = 256;  // Dn / Gn
constexpr int KP  = 68;   // padded f32 row stride for K LDS (bank spread)
constexpr int WP  = 264;  // padded f16 row stride for W LDS (bank spread)

__device__ __forceinline__ v8f wmma_f16(v16h a, v16h b, v8f c) {
  return __builtin_amdgcn_wmma_f32_16x16x32_f16(false, a, false, b, (short)0, c,
                                                false, false);
}

// A fragment (16x32 f16) from fp32 memory, converting on the fly.
// Lane L holds row M=L%16; halves 0..7 = K[i0..i0+7], halves 8..15 = K[i1..i1+7].
__device__ __forceinline__ v16h load_a_f32(const float* __restrict__ p, int i0, int i1) {
  v16h a;
#pragma unroll
  for (int i = 0; i < 8; ++i) a[i] = (_Float16)p[i0 + i];
#pragma unroll
  for (int i = 0; i < 8; ++i) a[8 + i] = (_Float16)p[i1 + i];
  return a;
}

// A fragment from f16 memory (LDS or global intermediate).
__device__ __forceinline__ v16h load_a_f16(const _Float16* __restrict__ p, int i0, int i1) {
  v16h a;
#pragma unroll
  for (int i = 0; i < 8; ++i) a[i] = p[i0 + i];
#pragma unroll
  for (int i = 0; i < 8; ++i) a[8 + i] = p[i1 + i];
  return a;
}

// B fragment (32x16 f16): lane L holds column N=L%16, halves j = K=(L/16)*16+j.
__device__ __forceinline__ v16h load_b_f16(const _Float16* __restrict__ p) {
  v16h b;
#pragma unroll
  for (int i = 0; i < 16; ++i) b[i] = p[i];
  return b;
}

// B fragment from 16 contiguous f32 (LDS-staged K), converting to f16.
__device__ __forceinline__ v16h load_b_f32(const float* p) {
  v16h b;
#pragma unroll
  for (int i = 0; i < 16; ++i) b[i] = (_Float16)p[i];
  return b;
}

// ---------------------------------------------------------------------------
// Kernel 1: flash attention. Block = 256 threads (8 waves); each wave owns a
// 16-row q tile; the block shares LDS-staged K/V chunks of 64 keys.
// K chunks are fetched with double-buffered GLOBAL_LOAD_ASYNC_TO_LDS_B128
// (ASYNCcnt-tracked) so HBM latency overlaps the WMMA work.
// Output: f16 attention result [B*SQ, D] written to workspace.
// ---------------------------------------------------------------------------
__global__ __launch_bounds__(256)
void GroupedCrossAttention_attn_kernel(const float* __restrict__ qg,
                                       const float* __restrict__ kg,
                                       const float* __restrict__ vg,
                                       const int*   __restrict__ kmask,
                                       const int*   __restrict__ qmask,
                                       _Float16*    __restrict__ ws) {
  __shared__ alignas(16) float    Ks32[2][64][KP];  // async-staged raw f32 K
  __shared__ alignas(16) _Float16 Vt[64][64];       // [hd][key] transposed f16
  __shared__ alignas(16) _Float16 Plds[8][16][64];  // per-wave P scratch
  __shared__ float kmaskS[64];

  const int tid  = threadIdx.x;
  const int wave = tid >> 5;
  const int lane = tid & 31;
  const int ln   = lane & 15;   // column / row-within-tile index
  const int hs   = lane >> 4;   // lane half select
  const int b    = blockIdx.z;
  const int h    = blockIdx.y;
  const int q0   = blockIdx.x * 128;

  // Issue the async K copy for one 64-key chunk into buffer nbuf.
  // 64 rows x 64 f32 = 16 KB -> 1024 x b128 -> 4 per thread.
  auto issue_k_async = [&](int kb, int nbuf) {
    const uint32_t lbase = (uint32_t)(uintptr_t)&Ks32[nbuf][0][0];
#pragma unroll
    for (int r = 0; r < 4; ++r) {
      const int idx = tid + 256 * r;
      const int key = idx >> 4;
      const int c4  = idx & 15;  // 16-byte chunk within the 64-float row
      const uint32_t laddr = lbase + (uint32_t)(key * KP + c4 * 4) * 4u;
      const float* gp =
          kg + ((size_t)b * SKn + kb + key) * Dn + h * HDn + c4 * 4;
      asm volatile("global_load_async_to_lds_b128 %0, %1, off"
                   :: "v"(laddr), "v"(gp)
                   : "memory");
    }
  };

  // Q fragments for this wave's 16 rows (kept in registers for whole kernel).
  const float* qrow =
      qg + ((size_t)b * SQn + q0 + wave * 16 + ln) * Dn + h * HDn;
  const v16h aq0 = load_a_f32(qrow, 0 + hs * 16, 8 + hs * 16);   // K = 0..31
  const v16h aq1 = load_a_f32(qrow, 32 + hs * 16, 40 + hs * 16);  // K = 32..63

  float mrun[8], lrun[8];
  v8f oacc[4];
#pragma unroll
  for (int j = 0; j < 8; ++j) { mrun[j] = -1e30f; lrun[j] = 0.0f; }
#pragma unroll
  for (int g = 0; g < 4; ++g) oacc[g] = v8f{};

  const float scl = 0.125f;  // 1/sqrt(HD=64)

  issue_k_async(/*kb=*/0, /*nbuf=*/0);  // prologue prefetch

  int buf = 0;
  for (int kb = 0; kb < SKn; kb += 64, buf ^= 1) {
    // Wait for this wave's async K copies (current buffer), then make all
    // waves' copies visible / separate from previous compute.
    asm volatile("s_wait_asynccnt 0" ::: "memory");
    __syncthreads();

    // Prefetch the NEXT K chunk into the other buffer while we work.
    if (kb + 64 < SKn) issue_k_async(kb + 64, buf ^ 1);

    // Synchronous V staging (transpose + f32->f16) and key-mask staging.
    for (int i = tid; i < 64 * 64; i += 256) {
      const int key = i >> 6;
      const int d   = i & 63;
      Vt[d][key] =
          (_Float16)vg[((size_t)b * SKn + kb + key) * Dn + h * HDn + d];
    }
    if (tid < 64)
      kmaskS[tid] = kmask[(size_t)b * SKn + kb + tid] ? 0.0f : -1e30f;
    __syncthreads();

    const float(*Kb)[KP] = Ks32[buf];

    // ---- S = Q * K^T (16 x 64), 8 WMMAs ----
    v8f sc[4];
#pragma unroll
    for (int t = 0; t < 4; ++t) {
      const int key   = t * 16 + ln;
      const float* kp = &Kb[key][hs * 16];
      const v16h b0 = load_b_f32(kp);        // head dims 0..31
      const v16h b1 = load_b_f32(kp + 32);   // head dims 32..63
      v8f s = v8f{};
      s = wmma_f16(aq0, b0, s);
      s = wmma_f16(aq1, b1, s);
      sc[t] = s;
    }

    float kmv[4];
#pragma unroll
    for (int t = 0; t < 4; ++t) kmv[t] = kmaskS[t * 16 + ln];

    // ---- online softmax over this 64-key chunk ----
    float rmax[8];
#pragma unroll
    for (int j = 0; j < 8; ++j) rmax[j] = -3.0e38f;
#pragma unroll
    for (int t = 0; t < 4; ++t)
#pragma unroll
      for (int j = 0; j < 8; ++j) {
        const float v = sc[t][j] * scl + kmv[t];
        sc[t][j] = v;
        rmax[j]  = fmaxf(rmax[j], v);
      }
#pragma unroll
    for (int msk = 1; msk < 16; msk <<= 1)
#pragma unroll
      for (int j = 0; j < 8; ++j)
        rmax[j] = fmaxf(rmax[j], __shfl_xor(rmax[j], msk, 32));

    float alpha[8], rs[8];
#pragma unroll
    for (int j = 0; j < 8; ++j) {
      const float nm = fmaxf(mrun[j], rmax[j]);
      alpha[j] = __expf(mrun[j] - nm);
      mrun[j]  = nm;
      rs[j]    = 0.0f;
    }
#pragma unroll
    for (int t = 0; t < 4; ++t)
#pragma unroll
      for (int j = 0; j < 8; ++j) {
        const float v = sc[t][j];
        const float p = (v < -1e29f) ? 0.0f : __expf(v - mrun[j]);
        sc[t][j] = p;
        rs[j] += p;
      }
#pragma unroll
    for (int msk = 1; msk < 16; msk <<= 1)
#pragma unroll
      for (int j = 0; j < 8; ++j) rs[j] += __shfl_xor(rs[j], msk, 32);
#pragma unroll
    for (int j = 0; j < 8; ++j) lrun[j] = lrun[j] * alpha[j] + rs[j];
#pragma unroll
    for (int g = 0; g < 4; ++g)
#pragma unroll
      for (int j = 0; j < 8; ++j) oacc[g][j] *= alpha[j];

    // ---- restripe P (C layout -> A layout) via per-wave LDS tile ----
    _Float16* pw = &Plds[wave][0][0];
#pragma unroll
    for (int t = 0; t < 4; ++t)
#pragma unroll
      for (int j = 0; j < 8; ++j)
        pw[(j + 8 * hs) * 64 + t * 16 + ln] = (_Float16)sc[t][j];
    // same-wave DS store->load dependency; compiler inserts s_wait_dscnt

    const _Float16* prow = pw + ln * 64;
    const v16h ap0 = load_a_f16(prow, 0 + hs * 16, 8 + hs * 16);   // keys 0..31
    const v16h ap1 = load_a_f16(prow, 32 + hs * 16, 40 + hs * 16);  // keys 32..63

    // ---- O += P * V (16 x 64), 8 WMMAs ----
#pragma unroll
    for (int g = 0; g < 4; ++g) {
      const int col = g * 16 + ln;
      const v16h bv0 = load_b_f16(&Vt[col][hs * 16]);
      const v16h bv1 = load_b_f16(&Vt[col][32 + hs * 16]);
      oacc[g] = wmma_f16(ap0, bv0, oacc[g]);
      oacc[g] = wmma_f16(ap1, bv1, oacc[g]);
    }
  }

  // ---- normalize, apply query mask, store f16 intermediate ----
#pragma unroll
  for (int j = 0; j < 8; ++j) {
    const int qrow_g = q0 + wave * 16 + j + 8 * hs;
    const int qmv    = qmask[(size_t)b * SQn + qrow_g];
    const float lv   = lrun[j];
    const float inv  = (qmv != 0 && lv > 0.0f) ? (1.0f / lv) : 0.0f;
    const size_t ob  = ((size_t)b * SQn + qrow_g) * Dn + h * HDn;
#pragma unroll
    for (int g = 0; g < 4; ++g)
      ws[ob + g * 16 + ln] = (_Float16)(oacc[g][j] * inv);
  }
}

// ---------------------------------------------------------------------------
// Kernel 2: grouped output projection + bias.
// out[n, g*GD + o] = sum_i ws[n, g*GD + i] * W[g, o, i] + bias[g*GD + o]
// One (group, 16-col) W slice is staged once per block into LDS (f16) and
// shared by all 8 waves, each computing a different 16-row tile (K=256 in
// 8 WMMA steps). Cuts W global traffic and conversion VALU by 8x.
// ---------------------------------------------------------------------------
__global__ __launch_bounds__(256)
void GroupedCrossAttention_proj_kernel(const _Float16* __restrict__ ws,
                                       const float*    __restrict__ wgt,
                                       const float*    __restrict__ bias,
                                       float*          __restrict__ out) {
  __shared__ alignas(16) _Float16 Wl[16][WP];  // [out-col local][i], padded

  const int tid  = threadIdx.x;
  const int wave = tid >> 5;
  const int lane = tid & 31;
  const int ln   = lane & 15;
  const int hs   = lane >> 4;

  const int ct  = blockIdx.x & 63;               // col tile (D/16 = 64 tiles)
  const int rt  = (blockIdx.x >> 6) * 8 + wave;  // row tile (N/16 = 256 tiles)
  const int g   = ct >> 4;
  const int oc0 = (ct & 15) * 16;                // first output col of tile

  // Stage W[g, oc0..oc0+15, 0..255] -> f16 LDS (shared by all 8 waves).
  for (int i = tid; i < 16 * GDn; i += 256) {
    const int r = i >> 8;     // local output column
    const int c = i & 255;    // input index
    Wl[r][c] = (_Float16)wgt[((size_t)g * GDn + oc0 + r) * GDn + c];
  }
  __syncthreads();

  const _Float16* ap = ws + (size_t)(rt * 16 + ln) * Dn + g * GDn;

  v8f acc = v8f{};
#pragma unroll
  for (int c0 = 0; c0 < GDn; c0 += 32) {
    const v16h a  = load_a_f16(ap, c0 + hs * 16, c0 + 8 + hs * 16);
    const v16h bw = load_b_f16(&Wl[ln][c0 + hs * 16]);
    acc = wmma_f16(a, bw, acc);
  }

  const float bv = bias[g * GDn + oc0 + ln];
#pragma unroll
  for (int j = 0; j < 8; ++j)
    out[(size_t)(rt * 16 + j + 8 * hs) * Dn + g * GDn + oc0 + ln] = acc[j] + bv;
}

extern "C" void kernel_launch(void* const* d_in, const int* in_sizes, int n_in,
                              void* d_out, int out_size, void* d_ws, size_t ws_size,
                              hipStream_t stream) {
  const float* q     = (const float*)d_in[0];
  const float* k     = (const float*)d_in[1];
  const float* v     = (const float*)d_in[2];
  const int*   kmask = (const int*)d_in[3];
  const int*   qmask = (const int*)d_in[4];
  const float* wgt   = (const float*)d_in[5];
  const float* bias  = (const float*)d_in[6];
  float*       out   = (float*)d_out;
  _Float16*    ws    = (_Float16*)d_ws;  // B*SQ*D f16 intermediate (8 MB)

  dim3 g1(SQn / 128, Hn, Bn);
  GroupedCrossAttention_attn_kernel<<<g1, 256, 0, stream>>>(q, k, v, kmask,
                                                            qmask, ws);

  // 2048 blocks: 64 col tiles x 32 row-tile groups (8 row tiles per block).
  GroupedCrossAttention_proj_kernel<<<2048, 256, 0, stream>>>(ws, wgt, bias,
                                                              out);
}
